// Block_89051851915272
// MI455X (gfx1250) — compile-verified, matrix operand order checked by reading, not run on hardware
//
#include <hip/hip_runtime.h>
#include <hip/hip_bf16.h>

// ---------------------------------------------------------------------------
// Transformer block for MI455X (gfx1250, wave32, WMMA bf16 -> f32 accum)
// GEMMs: 8-wave workgroup, 64x128 block tile, async global->LDS double buffer.
// ---------------------------------------------------------------------------

typedef __bf16 bf16;
typedef __attribute__((ext_vector_type(16))) __bf16 v16bf;
typedef __attribute__((ext_vector_type(8)))  float  v8f;
typedef int v4i __attribute__((vector_size(16)));

#define DIMC      1024
#define HEADS     16
#define HDIM      64
#define SEQ       2048
#define BATCH     2
#define NTOK      (BATCH * SEQ)      // 4096
#define HID       4096
#define THREE_D   3072
#define ATT_SCALE 0.125f             // 1/sqrt(64)

#if __has_builtin(__builtin_amdgcn_global_load_async_to_lds_b128) && \
    __has_builtin(__builtin_amdgcn_s_wait_asynccnt)
#define USE_ASYNC 1
#else
#define USE_ASYNC 0
#endif

typedef __attribute__((address_space(1))) v4i* gv4i_p;   // global int4*
typedef __attribute__((address_space(3))) v4i* lv4i_p;   // LDS int4*

union FragU { v16bf v; uint4 u[2]; };

// 16-byte global->LDS copy: async (ASYNCcnt) when available, else direct.
__device__ __forceinline__ void copy16_g2l(const bf16* __restrict__ g,
                                           bf16* __restrict__ l) {
#if USE_ASYNC
    __builtin_amdgcn_global_load_async_to_lds_b128((gv4i_p)g, (lv4i_p)l, 0, 0);
#else
    *(uint4*)l = *(const uint4*)g;
#endif
}

#if USE_ASYNC
#define WAIT_ASYNC(n) __builtin_amdgcn_s_wait_asynccnt(n)
#else
#define WAIT_ASYNC(n)
#endif

// Load a 16x32 bf16 WMMA fragment (A or B operand) from a row-major,
// K-contiguous matrix.  Per ISA 7.12.2: lane<16 holds row r, K {0..7} and
// {16..23}; lane>=16 holds row r, K {8..15} and {24..31}.
__device__ __forceinline__ v16bf load_frag_nt(const bf16* __restrict__ base,
                                              int ld, int row0, int k0, int lane) {
    const int half = lane >> 4;
    const int r    = lane & 15;
    const bf16* p  = base + (size_t)(row0 + r) * (size_t)ld + k0 + half * 8;
    FragU f;
    f.u[0] = *(const uint4*)p;
    f.u[1] = *(const uint4*)(p + 16);
    return f.v;
}

// Same fragment pattern but reading a staged LDS tile (k-window = 32 elems).
__device__ __forceinline__ v16bf lds_frag(const bf16* base, int ld,
                                          int row0, int lane) {
    const int half = lane >> 4;
    const int r    = lane & 15;
    const bf16* p  = base + (row0 + r) * ld + half * 8;
    FragU f;
    f.u[0] = *(const uint4*)p;
    f.u[1] = *(const uint4*)(p + 16);
    return f.v;
}

__device__ __forceinline__ float redmax16(float v) {
    v = fmaxf(v, __shfl_xor(v, 1, 32));
    v = fmaxf(v, __shfl_xor(v, 2, 32));
    v = fmaxf(v, __shfl_xor(v, 4, 32));
    v = fmaxf(v, __shfl_xor(v, 8, 32));
    return v;
}
__device__ __forceinline__ float redsum16(float v) {
    v += __shfl_xor(v, 1, 32);
    v += __shfl_xor(v, 2, 32);
    v += __shfl_xor(v, 4, 32);
    v += __shfl_xor(v, 8, 32);
    return v;
}

// ---------------------------------------------------------------------------
// fp32 -> bf16 conversion (weights)
// ---------------------------------------------------------------------------
__global__ void cvt_bf16_kernel(const float* __restrict__ in,
                                bf16* __restrict__ out, int n) {
    for (int i = blockIdx.x * blockDim.x + threadIdx.x; i < n;
         i += gridDim.x * blockDim.x)
        out[i] = (bf16)in[i];
}

// ---------------------------------------------------------------------------
// LayerNorm: one block per row of D=1024, eps = 0, f32 in -> bf16 out
// ---------------------------------------------------------------------------
__global__ void layernorm_kernel(const float* __restrict__ in,
                                 const float* __restrict__ w,
                                 bf16* __restrict__ out) {
    __shared__ float s_sum[256];
    __shared__ float s_sq[256];
    const int row = blockIdx.x;
    const float* x = in + (size_t)row * DIMC;

    float sum = 0.f, sq = 0.f;
    for (int c = threadIdx.x; c < DIMC; c += 256) {
        float v = x[c];
        sum += v;
        sq  += v * v;
    }
    s_sum[threadIdx.x] = sum;
    s_sq[threadIdx.x]  = sq;
    __syncthreads();
    for (int s = 128; s > 0; s >>= 1) {
        if (threadIdx.x < s) {
            s_sum[threadIdx.x] += s_sum[threadIdx.x + s];
            s_sq[threadIdx.x]  += s_sq[threadIdx.x + s];
        }
        __syncthreads();
    }
    const float mu   = s_sum[0] * (1.0f / DIMC);
    const float var  = s_sq[0] * (1.0f / DIMC) - mu * mu;
    const float rstd = rsqrtf(var);
    for (int c = threadIdx.x; c < DIMC; c += 256)
        out[(size_t)row * DIMC + c] = (bf16)((x[c] - mu) * rstd * w[c]);
}

// ---------------------------------------------------------------------------
// NT GEMM: C[M,N] = A[M,K] * W[N,K]^T  (A, W bf16 row-major, K-contiguous)
// 256 threads = 8 waves; block tile 64(M) x 128(N); BK=32; LDS double buffer
// filled with async global->LDS b128 copies (3 per thread per tile).
// Wave (wid&3) -> 16-row M sub-tile, (wid>>2) -> 64-col N sub-tile.
//   ep==0 : out_b = bf16(acc)
//   ep==1 : out_f = acc + res
//   ep==2 : out_b = bf16(gelu_exact(acc))
// ---------------------------------------------------------------------------
#define BM  64
#define BN  128
#define BK  32
#define LDT 40   // padded LDS row stride (elements) for a 32-element k-window

__global__ __launch_bounds__(256)
void gemm_nt_kernel(const bf16* __restrict__ A, const bf16* __restrict__ W,
                    int M, int N, int K,
                    const float* __restrict__ res,
                    float* __restrict__ out_f, bf16* __restrict__ out_b,
                    int ep) {
    __shared__ bf16 sA[2][BM * LDT];
    __shared__ bf16 sB[2][BN * LDT];

    const int tid  = threadIdx.x;
    const int lane = tid & 31;
    const int wid  = tid >> 5;
    const int half = lane >> 4;
    const int r    = lane & 15;

    const int n0 = blockIdx.x * BN;
    const int m0 = blockIdx.y * BM;
    const int wm = (wid & 3) * 16;   // wave M sub-tile within block
    const int wn = (wid >> 2) * 64;  // wave N sub-tile within block

    // Per-thread copy assignments (3 x 16B per tile: 1 for A, 2 for B).
    const int arow = tid >> 2, aseg = tid & 3;

    auto copy_tile = [&](int buf, int k0) {
        copy16_g2l(A + (size_t)(m0 + arow) * K + k0 + aseg * 8,
                   &sA[buf][arow * LDT + aseg * 8]);
#pragma unroll
        for (int j = 0; j < 2; ++j) {
            const int t2   = tid + j * 256;
            const int brow = t2 >> 2, bseg = t2 & 3;
            copy16_g2l(W + (size_t)(n0 + brow) * K + k0 + bseg * 8,
                       &sB[buf][brow * LDT + bseg * 8]);
        }
    };

    v8f acc[4] = {};

    copy_tile(0, 0);
    const int steps = K / BK;
    for (int ks = 0; ks < steps; ++ks) {
        const int cur = ks & 1;
        if (ks + 1 < steps) {
            copy_tile(cur ^ 1, (ks + 1) * BK);
            WAIT_ASYNC(3);   // drain current tile (in-order completion)
        } else {
            WAIT_ASYNC(0);
        }
        __syncthreads();

        v16bf a = lds_frag(sA[cur], LDT, wm, lane);
#pragma unroll
        for (int t = 0; t < 4; ++t) {
            v16bf b = lds_frag(sB[cur], LDT, wn + 16 * t, lane);
            acc[t] = __builtin_amdgcn_wmma_f32_16x16x32_bf16(
                false, a, false, b, (short)0, acc[t], false, false);
        }
        __syncthreads();   // all waves done reading before buffer is rewritten
    }

#pragma unroll
    for (int t = 0; t < 4; ++t) {
#pragma unroll
        for (int i = 0; i < 8; ++i) {
            const int row = m0 + wm + half * 8 + i;
            const int col = n0 + wn + 16 * t + r;
            const size_t idx = (size_t)row * (size_t)N + col;
            float v = acc[t][i];
            if (ep == 0) {
                out_b[idx] = (bf16)v;
            } else if (ep == 1) {
                out_f[idx] = v + res[idx];
            } else { // exact GELU
                float g = 0.5f * v * (1.0f + erff(v * 0.70710678118654752f));
                out_b[idx] = (bf16)g;
            }
        }
    }
}

// ---------------------------------------------------------------------------
// Repack fused QKV [tok, 3*1024] bf16 into per-head Q (pre-scaled), K, and
// V^T (d-major, L-contiguous so PV's B operand is K-contiguous).
// ---------------------------------------------------------------------------
__global__ void qkv_repack_kernel(const bf16* __restrict__ qkv,
                                  bf16* __restrict__ Q,
                                  bf16* __restrict__ Kb,
                                  bf16* __restrict__ Vt) {
    const int total = NTOK * THREE_D;
    for (int idx = blockIdx.x * blockDim.x + threadIdx.x; idx < total;
         idx += gridDim.x * blockDim.x) {
        const int row   = idx / THREE_D;
        const int e     = idx - row * THREE_D;
        const int which = e >> 10;           // 0=q 1=k 2=v
        const int rem   = e & 1023;
        const int h     = rem >> 6;
        const int d     = rem & 63;
        const int b     = row / SEQ;
        const int l     = row - b * SEQ;
        const int bh    = b * HEADS + h;
        const float v   = (float)qkv[idx];
        if (which == 0)
            Q[((size_t)bh * SEQ + l) * HDIM + d] = (bf16)(v * ATT_SCALE);
        else if (which == 1)
            Kb[((size_t)bh * SEQ + l) * HDIM + d] = (bf16)v;
        else
            Vt[((size_t)bh * HDIM + d) * SEQ + l] = (bf16)v;
    }
}

// ---------------------------------------------------------------------------
// Flash attention: one wave per (bh, 16-query block).  Per 32-key block:
// S = Q*K^T (4 WMMAs), online softmax (cross-lane row reductions), P bounced
// through LDS (C-layout -> A-layout), then P*V (4 WMMAs, d=64).
// ---------------------------------------------------------------------------
__global__ __launch_bounds__(32)
void flash_attn_kernel(const bf16* __restrict__ Q,
                       const bf16* __restrict__ Kb,
                       const bf16* __restrict__ Vt,
                       bf16* __restrict__ attn_out) {
    __shared__ bf16 Plds[16 * 32];

    const int lane = threadIdx.x & 31;
    const int half = lane >> 4;
    const int r    = lane & 15;
    const int q0   = blockIdx.x * 16;
    const int bh   = blockIdx.y;
    const int b    = bh / HEADS;
    const int h    = bh - b * HEADS;

    const bf16* Qbase = Q  + ((size_t)bh * SEQ + q0) * HDIM;
    const bf16* Kbase = Kb + (size_t)bh * SEQ * HDIM;
    const bf16* Vbase = Vt + (size_t)bh * HDIM * SEQ;

    const v16bf qf0 = load_frag_nt(Qbase, HDIM, 0, 0,  lane);
    const v16bf qf1 = load_frag_nt(Qbase, HDIM, 0, 32, lane);

    float mrow[8], lrow[8];
#pragma unroll
    for (int i = 0; i < 8; ++i) { mrow[i] = -3.0e38f; lrow[i] = 0.f; }
    v8f acc[4] = {};

    for (int key0 = 0; key0 < SEQ; key0 += 32) {
        // ---- S = Q * K^T  (two 16x16 column tiles) ----
        v8f s0 = {}, s1 = {};
        {
            v16bf kf = load_frag_nt(Kbase, HDIM, key0, 0, lane);
            s0 = __builtin_amdgcn_wmma_f32_16x16x32_bf16(false, qf0, false, kf,
                                                         (short)0, s0, false, false);
            kf = load_frag_nt(Kbase, HDIM, key0, 32, lane);
            s0 = __builtin_amdgcn_wmma_f32_16x16x32_bf16(false, qf1, false, kf,
                                                         (short)0, s0, false, false);
            kf = load_frag_nt(Kbase, HDIM, key0 + 16, 0, lane);
            s1 = __builtin_amdgcn_wmma_f32_16x16x32_bf16(false, qf0, false, kf,
                                                         (short)0, s1, false, false);
            kf = load_frag_nt(Kbase, HDIM, key0 + 16, 32, lane);
            s1 = __builtin_amdgcn_wmma_f32_16x16x32_bf16(false, qf1, false, kf,
                                                         (short)0, s1, false, false);
        }

        // ---- online softmax update over these 32 keys ----
#pragma unroll
        for (int i = 0; i < 8; ++i) {
            const float mij  = redmax16(fmaxf(s0[i], s1[i]));
            const float mnew = fmaxf(mrow[i], mij);
            const float alpha = __expf(mrow[i] - mnew);
            const float p0 = __expf(s0[i] - mnew);
            const float p1 = __expf(s1[i] - mnew);
            const float ls = redsum16(p0 + p1);
            lrow[i] = lrow[i] * alpha + ls;
            mrow[i] = mnew;
            Plds[(half * 8 + i) * 32 + r]      = (bf16)p0;
            Plds[(half * 8 + i) * 32 + 16 + r] = (bf16)p1;
            acc[0][i] *= alpha;
            acc[1][i] *= alpha;
            acc[2][i] *= alpha;
            acc[3][i] *= alpha;
        }
        __syncthreads();

        // ---- P (16x32) as A fragment from LDS ----
        FragU pf;
        {
            const bf16* p = Plds + r * 32 + half * 8;
            pf.u[0] = *(const uint4*)p;
            pf.u[1] = *(const uint4*)(p + 16);
        }
        __syncthreads();

        // ---- acc += P * V  (V^T rows = d-index, K-contiguous over keys) ----
#pragma unroll
        for (int t = 0; t < 4; ++t) {
            v16bf vf = load_frag_nt(Vbase, SEQ, t * 16, key0, lane);
            acc[t] = __builtin_amdgcn_wmma_f32_16x16x32_bf16(
                false, pf.v, false, vf, (short)0, acc[t], false, false);
        }
    }

    // ---- normalize and store [tok, D] bf16 (per-head columns) ----
#pragma unroll
    for (int i = 0; i < 8; ++i) {
        const float inv = 1.0f / lrow[i];
        const int grow = b * SEQ + q0 + half * 8 + i;
#pragma unroll
        for (int t = 0; t < 4; ++t) {
            const int col = h * HDIM + t * 16 + r;
            attn_out[(size_t)grow * DIMC + col] = (bf16)(acc[t][i] * inv);
        }
    }
}

// ---------------------------------------------------------------------------
// Host-side launch
// ---------------------------------------------------------------------------
extern "C" void kernel_launch(void* const* d_in, const int* in_sizes, int n_in,
                              void* d_out, int out_size, void* d_ws, size_t ws_size,
                              hipStream_t stream) {
    const float* x      = (const float*)d_in[0];
    const float* ln1_w  = (const float*)d_in[1];
    const float* ln2_w  = (const float*)d_in[2];
    const float* qkv_w  = (const float*)d_in[3];
    const float* proj_w = (const float*)d_in[4];
    const float* mlp_w1 = (const float*)d_in[5];
    const float* mlp_w2 = (const float*)d_in[6];
    float* out = (float*)d_out;

    // Workspace carve-up (all sizes multiples of 256B).
    char* p = (char*)d_ws;
    auto take = [&](size_t bytes) { char* q = p; p += (bytes + 255) & ~(size_t)255; return q; };

    bf16* wq_b   = (bf16*)take((size_t)THREE_D * DIMC * 2);   // qkv_w  bf16
    bf16* wp_b   = (bf16*)take((size_t)DIMC * DIMC * 2);      // proj_w bf16
    bf16* w1_b   = (bf16*)take((size_t)HID  * DIMC * 2);      // mlp_w1 bf16
    bf16* w2_b   = (bf16*)take((size_t)DIMC * HID  * 2);      // mlp_w2 bf16
    bf16* ln1_b  = (bf16*)take((size_t)NTOK * DIMC * 2);      // LN1(x) bf16
    bf16* qkv_b  = (bf16*)take((size_t)NTOK * THREE_D * 2);   // fused qkv bf16
    bf16* Qb     = (bf16*)take((size_t)BATCH * HEADS * SEQ * HDIM * 2);
    bf16* Kb     = (bf16*)take((size_t)BATCH * HEADS * SEQ * HDIM * 2);
    bf16* Vt     = (bf16*)take((size_t)BATCH * HEADS * HDIM * SEQ * 2);
    bf16* att_b  = (bf16*)take((size_t)NTOK * DIMC * 2);      // attn out bf16
    float* Xres  = (float*)take((size_t)NTOK * DIMC * 4);     // x + attn proj
    bf16* y2_b   = (bf16*)take((size_t)NTOK * DIMC * 2);      // LN2(X) bf16
    bf16* h_b    = (bf16*)take((size_t)NTOK * HID * 2);       // gelu(mlp1) bf16
    (void)ws_size; (void)in_sizes; (void)n_in; (void)out_size;

    // 1. weights -> bf16
    cvt_bf16_kernel<<<2048, 256, 0, stream>>>(qkv_w,  wq_b, THREE_D * DIMC);
    cvt_bf16_kernel<<<2048, 256, 0, stream>>>(proj_w, wp_b, DIMC * DIMC);
    cvt_bf16_kernel<<<2048, 256, 0, stream>>>(mlp_w1, w1_b, HID * DIMC);
    cvt_bf16_kernel<<<2048, 256, 0, stream>>>(mlp_w2, w2_b, DIMC * HID);

    // 2. LN1
    layernorm_kernel<<<NTOK, 256, 0, stream>>>(x, ln1_w, ln1_b);

    // 3. QKV GEMM: [4096,1024] x [3072,1024]^T -> bf16
    gemm_nt_kernel<<<dim3(THREE_D / BN, NTOK / BM), 256, 0, stream>>>(
        ln1_b, wq_b, NTOK, THREE_D, DIMC, nullptr, nullptr, qkv_b, 0);

    // 4. split/scale/transpose into Q, K, V^T
    qkv_repack_kernel<<<4096, 256, 0, stream>>>(qkv_b, Qb, Kb, Vt);

    // 5. flash attention: grid (128 query blocks, 32 bh)
    flash_attn_kernel<<<dim3(SEQ / 16, BATCH * HEADS), 32, 0, stream>>>(
        Qb, Kb, Vt, att_b);

    // 6. proj GEMM + residual: X = x + att @ proj_w^T
    gemm_nt_kernel<<<dim3(DIMC / BN, NTOK / BM), 256, 0, stream>>>(
        att_b, wp_b, NTOK, DIMC, DIMC, x, Xres, nullptr, 1);

    // 7. LN2
    layernorm_kernel<<<NTOK, 256, 0, stream>>>(Xres, ln2_w, y2_b);

    // 8. MLP up + exact GELU -> bf16
    gemm_nt_kernel<<<dim3(HID / BN, NTOK / BM), 256, 0, stream>>>(
        y2_b, w1_b, NTOK, HID, DIMC, nullptr, nullptr, h_b, 2);

    // 9. MLP down + residual -> d_out (f32)
    gemm_nt_kernel<<<dim3(DIMC / BN, NTOK / BM), 256, 0, stream>>>(
        h_b, w2_b, NTOK, DIMC, HID, Xres, out, nullptr, 1);
}
